// NodeModel_38714835206272
// MI455X (gfx1250) — matrix operand order repaired and checked by reading, not compiled, not run on hardware
//
#include <hip/hip_runtime.h>
#include <hip/hip_bf16.h>

typedef __attribute__((ext_vector_type(16))) _Float16 v16h;
typedef __attribute__((ext_vector_type(8)))  float    v8f;

#define LRELU_ALPHA 0.2f

// ---- order-preserving float<->uint map (for atomic segment-max) ----
__device__ __forceinline__ unsigned ord_of(float f) {
    unsigned u = __float_as_uint(f);
    return (u & 0x80000000u) ? ~u : (u | 0x80000000u);
}
__device__ __forceinline__ float float_of_ord(unsigned o) {
    return (o & 0x80000000u) ? __uint_as_float(o & 0x7FFFFFFFu)
                             : __uint_as_float(~o);
}

// ---- utility kernels ----
__global__ void k_zero_f32(float* p, long long n) {
    long long i = (long long)blockIdx.x * blockDim.x + threadIdx.x;
    if (i < n) p[i] = 0.0f;
}

// ---- pack B (f32 [K,Ncols], row major) into WMMA fragment order (f16) ----
// Fragment f = (kt*NT + tn): 32 lanes x 16 halves, contiguous 32B per lane.
// Lane layout (this code's contiguous k order, matching A; not ISA table order): lanes 0-15: N=lane, K=kt*32+0..15;
// lanes 16-31: N=lane-16, K=kt*32+16..31; element j <-> K offset j.
__global__ void k_pack_b(const float* __restrict__ W, _Float16* __restrict__ Bp,
                         int K, int Ncols) {
    int i = blockIdx.x * blockDim.x + threadIdx.x;   // (kt*NT + tn)*32 + lane
    const int NT = Ncols / 16;
    const int total = (K / 32) * NT * 32;
    if (i >= total) return;
    int lane = i & 31;
    int tn   = (i >> 5) % NT;
    int kt   = (i >> 5) / NT;
    int l15 = lane & 15, hi = lane >> 4;
    int kbase = kt * 32 + (hi ? 16 : 0);
    int col = tn * 16 + l15;
    _Float16* dst = Bp + (size_t)i * 16;
    #pragma unroll
    for (int j = 0; j < 16; ++j)
        dst[j] = (_Float16)W[(size_t)(kbase + j) * Ncols + col];
}

// ---- WMMA GEMM: C[M,Ncols] = act(A[M,K]) * B, f16 inputs / f32 accum.
// One wave computes a full 16-row x Ncols stripe (NT = Ncols/16 accumulators),
// so A is streamed exactly once. B is pre-packed in fragment order; all NT
// fragments of a K-step are loaded up-front so the WMMA chain drains the
// load counter progressively instead of stalling per instruction.
template<int NT, bool ELU, int WAVES>
__global__ __launch_bounds__(32 * WAVES)
void k_gemm_wmma(const float* __restrict__ A, const _Float16* __restrict__ Bp,
                 float* __restrict__ C, int M, int K, int Ncols) {
    const int lane = threadIdx.x & 31;
    const int wave = threadIdx.x >> 5;
    const int tm = blockIdx.x * WAVES + wave;
    if (tm * 16 >= M) return;                    // uniform per wave
    const int l15 = lane & 15, hi = lane >> 4;

    int row  = tm * 16 + l15;
    int rowc = row < M ? row : (M - 1);          // clamp: EXEC stays all-ones
    const float* Arow = A + (size_t)rowc * K;

    v8f acc[NT] = {};
    const int KT = K / 32;
    for (int kt = 0; kt < KT; ++kt) {
        // Issue all B-fragment loads for this K-step first (one clause).
        v16h b[NT];
        #pragma unroll
        for (int tn = 0; tn < NT; ++tn)
            b[tn] = *(const v16h*)(Bp + ((size_t)(kt * NT + tn) * 32 + lane) * 16);

        // A fragment: element j = K (kt*32 + kbase + j) for j<8,
        //             element 8+j = K (kt*32 + kbase + 8 + j); kbase = hi?16:0.
        const int ka = kt * 32 + (hi ? 16 : 0);
        v16h a;
        #pragma unroll
        for (int j = 0; j < 8; ++j) {
            float f0 = Arow[ka + j];
            float f1 = Arow[ka + 8 + j];
            if (ELU) {
                f0 = f0 > 0.0f ? f0 : (expf(f0) - 1.0f);
                f1 = f1 > 0.0f ? f1 : (expf(f1) - 1.0f);
            }
            a[j]     = (_Float16)f0;
            a[8 + j] = (_Float16)f1;
        }
        #pragma unroll
        for (int tn = 0; tn < NT; ++tn)
            acc[tn] = __builtin_amdgcn_wmma_f32_16x16x32_f16(
                          false, a, false, b[tn], (short)0, acc[tn], false, false);
    }
    // C/D layout: VGPR r -> M = r (lanes 0-15) / 8+r (lanes 16-31); N = lane&15.
    if (tm * 16 + 16 <= M) {                     // fast path: no predication
        #pragma unroll
        for (int tn = 0; tn < NT; ++tn) {
            #pragma unroll
            for (int r = 0; r < 8; ++r) {
                int m = tm * 16 + (hi ? (8 + r) : r);
                C[(size_t)m * Ncols + tn * 16 + l15] = acc[tn][r];
            }
        }
    } else {                                     // tail tile
        for (int tn = 0; tn < NT; ++tn)
            for (int r = 0; r < 8; ++r) {
                int m = tm * 16 + (hi ? (8 + r) : r);
                if (m < M) C[(size_t)m * Ncols + tn * 16 + l15] = acc[tn][r];
            }
    }
}

// ---- per-node attention features: f_src/f_dst dots ----
template<int H, int D>
__global__ void k_feats(const float* __restrict__ hbuf, const float* __restrict__ a,
                        float* __restrict__ fs, float* __restrict__ fd, long long N) {
    long long i = (long long)blockIdx.x * blockDim.x + threadIdx.x; // n*H + h
    if (i >= N * H) return;
    int h = (int)(i % H);
    long long n = i / H;
    const float* hv = hbuf + n * (long long)(H * D) + (long long)h * D;
    const float* av = a + (long long)h * 2 * D;
    float s = 0.0f, d = 0.0f;
    #pragma unroll
    for (int k = 0; k < D; ++k) {
        float x = hv[k];
        s += x * av[k];
        d += x * av[D + k];
    }
    fs[i] = s;
    fd[i] = d;
}

// ---- edge pass 1: segment max of leaky-relu logits (atomic on ordered uints)
// One thread per EDGE; per-head loads are consecutive -> b128 gathers.
template<int H>
__global__ void k_edge_max(const long long* __restrict__ edges, long long E,
                           const float* __restrict__ fs, const float* __restrict__ fd,
                           unsigned* __restrict__ m) {
    long long e = (long long)blockIdx.x * blockDim.x + threadIdx.x;
    if (e >= E) return;
    long long s = edges[e], d = edges[E + e];
    const float* fsp = fs + s * H;
    const float* fdp = fd + d * H;
    unsigned*    mp  = m  + d * H;
    #pragma unroll
    for (int h = 0; h < H; ++h) {
        float v = fsp[h] + fdp[h];
        v = v > 0.0f ? v : LRELU_ALPHA * v;
        atomicMax(&mp[h], ord_of(v));
    }
}

// ---- edge pass 2: segment sum of exp(e - max); one thread per edge
template<int H>
__global__ void k_edge_sum(const long long* __restrict__ edges, long long E,
                           const float* __restrict__ fs, const float* __restrict__ fd,
                           const unsigned* __restrict__ m, float* __restrict__ ssum) {
    long long e = (long long)blockIdx.x * blockDim.x + threadIdx.x;
    if (e >= E) return;
    long long s = edges[e], d = edges[E + e];
    const float*    fsp = fs + s * H;
    const float*    fdp = fd + d * H;
    const unsigned* mp  = m  + d * H;
    float*          sp  = ssum + d * H;
    #pragma unroll
    for (int h = 0; h < H; ++h) {
        float v = fsp[h] + fdp[h];
        v = v > 0.0f ? v : LRELU_ALPHA * v;
        atomicAdd(&sp[h], expf(v - float_of_ord(mp[h])));
    }
}

// ---- edge pass 3: scatter att * h[src] into out[dst]; thread per (edge, head)
template<int H, int D>
__global__ void k_edge_aggr(const long long* __restrict__ edges, long long E,
                            const float* __restrict__ fs, const float* __restrict__ fd,
                            const unsigned* __restrict__ m, const float* __restrict__ ssum,
                            const float* __restrict__ hbuf, float* __restrict__ out) {
    long long i = (long long)blockIdx.x * blockDim.x + threadIdx.x;
    if (i >= E * H) return;
    int h = (int)(i % H);
    long long e = i / H;
    long long s = edges[e], d = edges[E + e];
    float v = fs[s * H + h] + fd[d * H + h];
    v = v > 0.0f ? v : LRELU_ALPHA * v;
    float mx  = float_of_ord(m[d * H + h]);
    float att = expf(v - mx) / (ssum[d * H + h] + 1e-16f);
    const float* hs = hbuf + s * (long long)(H * D) + (long long)h * D;
    float*       od = out  + d * (long long)(H * D) + (long long)h * D;
    #pragma unroll
    for (int k = 0; k < D; ++k) atomicAdd(&od[k], att * hs[k]);
}

extern "C" void kernel_launch(void* const* d_in, const int* in_sizes, int n_in,
                              void* d_out, int out_size, void* d_ws, size_t ws_size,
                              hipStream_t stream) {
    (void)n_in; (void)out_size; (void)ws_size;
    // Reference dims: d_input=256, H=8, d_hidden=16, nclasses=16
    const int DIN = 256, H = 8, DH = 16, C = 16, HD = H * DH; // HD = 128
    const long long N = (long long)in_sizes[0] / DIN;
    const long long E = (long long)in_sizes[1] / 2;

    const float*     x     = (const float*)d_in[0];
    const long long* edges = (const long long*)d_in[1];
    const float*     W1    = (const float*)d_in[2];
    const float*     a1    = (const float*)d_in[3];
    const float*     W2    = (const float*)d_in[4];
    const float*     a2    = (const float*)d_in[5];
    float* out2 = (float*)d_out; // [N, C]

    // workspace carve-up (out1|m1|s1 contiguous -> single zero; m2|s2 likewise)
    char* ws = (char*)d_ws;
    float*    h1   = (float*)ws;                 ws += (size_t)N * HD * 4;
    float*    out1 = (float*)ws;                 ws += (size_t)N * HD * 4;
    unsigned* m1   = (unsigned*)ws;              ws += (size_t)N * H * 4;
    float*    s1   = (float*)ws;                 ws += (size_t)N * H * 4;
    float*    f1s  = (float*)ws;                 ws += (size_t)N * H * 4;
    float*    f1d  = (float*)ws;                 ws += (size_t)N * H * 4;
    float*    h2   = (float*)ws;                 ws += (size_t)N * C * 4;
    unsigned* m2   = (unsigned*)ws;              ws += (size_t)N * 4;
    float*    s2   = (float*)ws;                 ws += (size_t)N * 4;
    float*    f2s  = (float*)ws;                 ws += (size_t)N * 4;
    float*    f2d  = (float*)ws;                 ws += (size_t)N * 4;
    _Float16* Bp1  = (_Float16*)ws;              ws += (size_t)DIN * HD * 2; // 32B aligned
    _Float16* Bp2  = (_Float16*)ws;

    const int B = 256;
    auto blocks = [](long long n, int b) { return (unsigned)((n + b - 1) / b); };

    // pack weights into WMMA fragment order (f32 -> f16)
    k_pack_b<<<blocks(DIN * HD, B), B, 0, stream>>>(W1, Bp1, DIN, HD);
    k_pack_b<<<blocks(HD * C, B), B, 0, stream>>>(W2, Bp2, HD, C);

    // ===== layer 1 =====
    // zero out1 + m1 + s1 in one sweep (0u is ord-max identity)
    k_zero_f32<<<blocks(N * (HD + 2 * H), B), B, 0, stream>>>(out1, N * (HD + 2 * H));

    const int WAVES = 4;
    unsigned tiles = (unsigned)((N + 15) / 16);
    k_gemm_wmma<8, false, WAVES><<<(tiles + WAVES - 1) / WAVES, 32 * WAVES, 0, stream>>>(
        x, Bp1, h1, (int)N, DIN, HD);

    k_feats<8, 16><<<blocks(N * H, B), B, 0, stream>>>(h1, a1, f1s, f1d, N);
    k_edge_max<8><<<blocks(E, B), B, 0, stream>>>(edges, E, f1s, f1d, m1);
    k_edge_sum<8><<<blocks(E, B), B, 0, stream>>>(edges, E, f1s, f1d, m1, s1);
    k_edge_aggr<8, 16><<<blocks(E * H, B), B, 0, stream>>>(edges, E, f1s, f1d, m1, s1, h1, out1);

    // ===== layer 2 (1 head, C dims); ELU fused into the GEMM's A conversion =====
    k_gemm_wmma<1, true, WAVES><<<(tiles + WAVES - 1) / WAVES, 32 * WAVES, 0, stream>>>(
        out1, Bp2, h2, (int)N, HD, C);

    k_feats<1, 16><<<blocks(N, B), B, 0, stream>>>(h2, a2, f2s, f2d, N);
    // zero m2 + s2 in one sweep; zero d_out accumulator
    k_zero_f32<<<blocks(2 * N, B), B, 0, stream>>>((float*)m2, 2 * N);
    k_zero_f32<<<blocks(N * C, B), B, 0, stream>>>(out2, N * C);

    k_edge_max<1><<<blocks(E, B), B, 0, stream>>>(edges, E, f2s, f2d, m2);
    k_edge_sum<1><<<blocks(E, B), B, 0, stream>>>(edges, E, f2s, f2d, m2, s2);
    k_edge_aggr<1, 16><<<blocks(E, B), B, 0, stream>>>(edges, E, f2s, f2d, m2, s2, h2, out2);
    // mean over 1 head == identity, out2 (= d_out) is the final result
}
